// GraphConv_39256001085744
// MI455X (gfx1250) — matrix operand order repaired and checked by reading, not compiled, not run on hardware
//
#include <hip/hip_runtime.h>
#include <cstdint>
#include <cstddef>

// ---------------------------------------------------------------------------
// Graph diffusion conv for MI455X (gfx1250, wave32, WMMA bf16).
//   B=64, C=64, N=512, T=12, E=2, ORDER=2, C_cat=320, C_out=64
//
// Activations use a "k-pair interleaved" bf16 layout:
//   Xpk[b][v>>1][n*2 + (v&1)],  n = c*12 + t
// so async copies of a 32(v) x 128(n) tile land in LDS already dword-packed
// for the WMMA B-operand fragments (each frag reg = one ds_load_b32).
//
// Data movement per K-tile of the diffusion GEMM:
//   - A tile (128x32 of AT): TENSOR_LOAD_TO_LDS (TDM), issued by wave 0,
//     pipelined with s_wait_tensorcnt.
//   - B tile (16 pair-rows x 512B): GLOBAL_LOAD_ASYNC_TO_LDS_B128 per lane,
//     pipelined with s_wait_asynccnt.
//   Double-buffered LDS, workgroup barriers publish tiles.
// ---------------------------------------------------------------------------

typedef __bf16        v16bf __attribute__((ext_vector_type(16)));
typedef float         v8f   __attribute__((ext_vector_type(8)));
typedef unsigned int  v8u   __attribute__((ext_vector_type(8)));
typedef unsigned int  u32x4 __attribute__((ext_vector_type(4)));
typedef int           i32x4 __attribute__((ext_vector_type(4)));
typedef int           i32x8 __attribute__((ext_vector_type(8)));

union FragU { v8u u; v16bf b; };

__device__ __forceinline__ uint16_t f2bf(float f) {
  union { float f; uint32_t u; } v; v.f = f;
  uint32_t r = v.u + 0x7FFFu + ((v.u >> 16) & 1u);   // round-to-nearest-even
  return (uint16_t)(r >> 16);
}

__device__ __forceinline__ uint32_t lds_addr_of(const void* p) {
  return (uint32_t)(uintptr_t)p;   // generic LDS pointer: low dword = LDS offset
}

// Async global->LDS copy, 16B per lane (tracked by ASYNCcnt).
__device__ __forceinline__ void async_copy_b128(uint32_t lds_addr,
                                                const uint16_t* gaddr) {
  asm volatile("global_load_async_to_lds_b128 %0, %1, off"
               :: "v"(lds_addr), "v"(gaddr) : "memory");
}
__device__ __forceinline__ void async_copy_b128_off16(uint32_t lds_addr,
                                                      const uint16_t* gaddr) {
  asm volatile("global_load_async_to_lds_b128 %0, %1, off offset:16"
               :: "v"(lds_addr), "v"(gaddr) : "memory");
}

// TDM: DMA a 2-D bf16 tile (tile_dim0 x tile_dim1 elements) from a row-major
// tensor (row length = stride0 elements) into contiguous LDS. Tracked by
// TENSORcnt. D# layout per CDNA5 ISA ch.8 (group0: count/lds/global/type,
// group1: data_size, tensor dims, tile dims, strides; groups 2/3 unused=0).
__device__ __forceinline__ void tdm_load_tile_2d(uint32_t lds_byte_addr,
                                                 const uint16_t* gtile,
                                                 uint32_t tensor_dim0,
                                                 uint32_t tensor_dim1,
                                                 uint32_t stride0,
                                                 uint32_t tile_dim0,
                                                 uint32_t tile_dim1) {
  const uint64_t ga = (uint64_t)(uintptr_t)gtile;          // 57-bit byte addr
  u32x4 g0;
  g0[0] = 1u;                                              // count=1 (valid D#)
  g0[1] = lds_byte_addr;                                   // lds_addr  [63:32]
  g0[2] = (uint32_t)ga;                                    // gaddr     [95:64]
  g0[3] = (uint32_t)((ga >> 32) & 0x01FFFFFFu) | (2u << 30); // gaddr hi | type=2
  i32x8 g1;
  g1[0] = (int)(1u << 16);                                 // data_size=1 (2B)
  g1[1] = (int)(tensor_dim0 << 16);                        // dim0 low16 @ [63:48]
  g1[2] = (int)((tensor_dim0 >> 16) | (tensor_dim1 << 16));// dim0 hi | dim1 low
  g1[3] = (int)((tensor_dim1 >> 16) | (tile_dim0 << 16));  // dim1 hi | tile0
  g1[4] = (int)tile_dim1;                                  // tile1 | tile2=0
  g1[5] = (int)stride0;                                    // dim0_stride low32
  g1[6] = 0;                                               // stride hi | d1s lo
  g1[7] = 0;
  const i32x4 z4 = {0, 0, 0, 0};
#if __has_include(<hip/amd_detail/amd_gfx1250_TDM.h>)
  const i32x8 z8 = {0, 0, 0, 0, 0, 0, 0, 0};
  __builtin_amdgcn_tensor_load_to_lds(g0, g1, z4, z4, z8, 0);
#else
  __builtin_amdgcn_tensor_load_to_lds(g0, g1, z4, z4, 0);
#endif
}

#define BATCH 64
#define NND   512
#define CT    768      // C*T = 64*12
#define CT2   1536     // interleaved pair-row length (2*CT)
#define TT    12

// -------------------------------------------------------------------
// 1) x [B,C,N,T] f32 -> Xpk (pair-interleaved bf16), coalesced reads
// -------------------------------------------------------------------
__global__ __launch_bounds__(256)
void convert_x_kernel(const float* __restrict__ x, uint16_t* __restrict__ Xpk) {
  int idx = blockIdx.x * 256 + threadIdx.x;           // enumerates (b,c,v,t)
  int t = idx % TT;
  int v = (idx / TT) & (NND - 1);
  int c = (idx / (TT * NND)) & 63;
  int b = idx / (TT * NND * 64);
  float val = x[idx];
  const int n = c * TT + t;
  Xpk[(size_t)b * NND * CT + (size_t)(v >> 1) * CT2 + n * 2 + (v & 1)] = f2bf(val);
}

// -------------------------------------------------------------------
// 2) supports [E,N,N] f32 -> AT [E,N,N] bf16 transposed; W f32 -> bf16
// -------------------------------------------------------------------
__global__ __launch_bounds__(256)
void convert_aw_kernel(const float* __restrict__ adj, const float* __restrict__ W,
                       uint16_t* __restrict__ AT, uint16_t* __restrict__ Wb) {
  int idx = blockIdx.x * 256 + threadIdx.x;
  if (idx < 2 * NND * NND) {
    int e = idx >> 18;
    int v = (idx >> 9) & (NND - 1);
    int w = idx & (NND - 1);
    AT[(e << 18) + w * NND + v] = f2bf(adj[idx]);     // AT[e][w][v] = adj[e][v][w]
  } else {
    int j = idx - 2 * NND * NND;                      // j < 64*320 = 20480
    Wb[j] = f2bf(W[j]);
  }
}

// -------------------------------------------------------------------
// 3) Diffusion GEMM:  Y[b] = AT (512x512) @ X[b] (512x768), bf16->bf16
//    Block tile 128(M) x 128(N), 256 threads = 8 waves, each wave 2x4
//    16x16 subtiles. A tile via TDM (wave 0), B tile via per-lane async
//    (2 instr/thread/tile). Double-buffered LDS.
// -------------------------------------------------------------------
__global__ __launch_bounds__(256)
void diffusion_gemm_kernel(const uint16_t* __restrict__ AT,   // [512][512]
                           const uint16_t* __restrict__ Xpk,  // [B] pair-interleaved
                           uint16_t* __restrict__ Ypk)        // [B] pair-interleaved
{
  __shared__ uint16_t As[2][128 * 32];   // [buf][m][k]   raw bf16 rows (8KB)
  __shared__ uint16_t Bs[2][16 * 256];   // [buf][kp][n*2+par] packed pairs (8KB)

  const int ntile = blockIdx.x;       // 0..5
  const int mtile = blockIdx.y;       // 0..3
  const int b     = blockIdx.z;       // 0..63
  const int tid   = threadIdx.x;
  const int lane  = tid & 31;
  const int wave  = tid >> 5;         // 0..7
  const int mw    = wave & 3;         // M quarter  (2 subtiles)
  const int nw    = wave >> 2;        // N half     (4 subtiles)
  const int half  = lane >> 4;
  const int l15   = lane & 15;

  const uint16_t* Xb = Xpk + (size_t)b * NND * CT;
  uint16_t*       Yb = Ypk + (size_t)b * NND * CT;

  v8f acc[2][4];
#pragma unroll
  for (int i = 0; i < 2; ++i)
#pragma unroll
    for (int j = 0; j < 4; ++j)
      acc[i][j] = (v8f){0.f,0.f,0.f,0.f,0.f,0.f,0.f,0.f};

  // A tile: one TDM descriptor per tile, issued by wave 0 only.
  auto issue_a = [&](int buf, int k0) {
    tdm_load_tile_2d(lds_addr_of(&As[buf][0]),
                     AT + (size_t)(mtile * 128) * NND + k0,
                     /*tensor_dim0=*/NND, /*tensor_dim1=*/NND,
                     /*stride0=*/NND, /*tile_dim0=*/32, /*tile_dim1=*/128);
  };
  // B tile: 16 pair-rows x 512B; each thread copies 32B (2 x b128 async).
  auto issue_b = [&](int buf, int k0) {
    const int kp  = tid >> 4;            // pair-row 0..15
    const int seg = (tid & 15) * 16;     // u16 elements within 256-element (128-dword) row
    const uint16_t* g = Xb + (size_t)((k0 >> 1) + kp) * CT2 + ntile * 256 + seg;
    const uint32_t  l = lds_addr_of(&Bs[buf][kp * 256 + seg]);
    async_copy_b128(l, g);
    async_copy_b128_off16(l + 16, g);
  };

  if (wave == 0) issue_a(0, 0);
  issue_b(0, 0);

  for (int ks = 0; ks < NND / 32; ++ks) {
    const int cur = ks & 1;
    if (ks + 1 < NND / 32) {
      if (wave == 0) issue_a(cur ^ 1, (ks + 1) * 32);
      issue_b(cur ^ 1, (ks + 1) * 32);
      asm volatile("s_wait_asynccnt 0x2" ::: "memory");   // B tile ks complete
      if (wave == 0) __builtin_amdgcn_s_wait_tensorcnt(1); // A tile ks complete
    } else {
      asm volatile("s_wait_asynccnt 0x0" ::: "memory");
      if (wave == 0) __builtin_amdgcn_s_wait_tensorcnt(0);
    }
    __syncthreads();   // publish tile ks to all waves

    // ---- fragments (contiguous k order: element s of half h holds k = 16h + s in A and B, wave32) ----
    FragU afr[2], bfr[4];
#pragma unroll
    for (int i = 0; i < 2; ++i) {
      const int m = (mw * 2 + i) * 16 + l15;
#pragma unroll
      for (int r = 0; r < 8; ++r) {
        const int kl = half * 16 + r * 2;   // even
        afr[i].u[r] = *(const uint32_t*)&As[cur][m * 32 + kl];
      }
    }
    const uint32_t* Bs32 = (const uint32_t*)&Bs[cur][0];   // [kp][n] dwords
#pragma unroll
    for (int j = 0; j < 4; ++j) {
      const int n = (nw * 4 + j) * 16 + l15;
#pragma unroll
      for (int r = 0; r < 8; ++r)
        bfr[j].u[r] = Bs32[(half * 8 + r) * 128 + n];
    }

#pragma unroll
    for (int i = 0; i < 2; ++i)
#pragma unroll
      for (int j = 0; j < 4; ++j)
        acc[i][j] = __builtin_amdgcn_wmma_f32_16x16x32_bf16(
            false, afr[i].b, false, bfr[j].b, (short)0, acc[i][j], false, false);

    __syncthreads();   // done reading buf `cur` before it is refilled
  }

  // ---- epilogue: C layout VGPR r -> M = half*8 + r, N = lane&15 ----
  // Write into the pair-interleaved layout: Y[w][n] -> Yb[(w>>1)*CT2 + n*2 + (w&1)]
#pragma unroll
  for (int i = 0; i < 2; ++i)
#pragma unroll
    for (int j = 0; j < 4; ++j)
#pragma unroll
      for (int r = 0; r < 8; ++r) {
        const int w = mtile * 128 + (mw * 2 + i) * 16 + half * 8 + r;
        const int n = ntile * 128 + (nw * 4 + j) * 16 + l15;
        Yb[(size_t)(w >> 1) * CT2 + n * 2 + (w & 1)] = f2bf(acc[i][j][r]);
      }
}

// -------------------------------------------------------------------
// 4) Projection: per (b, v-pair): out[64x12] = W(64x320) @ H(320x12) + bias
//    H = 5 concatenated pair-interleaved bf16 blocks (contiguous in ws).
//    256 threads = 8 waves: waves 0-3 -> parity 0, waves 4-7 -> parity 1;
//    within a parity, wave w handles output rows o = 16w..16w+15.
// -------------------------------------------------------------------
__global__ __launch_bounds__(256)
void proj_gemm_kernel(const uint16_t* __restrict__ H,   // 5 * [B] pair-interleaved
                      const uint16_t* __restrict__ Wb,  // [64][320]
                      const float* __restrict__ bias,   // [64]
                      float* __restrict__ out)          // [B][64][512][12]
{
  __shared__ uint16_t Hs[5 * CT2 + 32];  // [f][t] interleaved pair + tail pad

  const int bvp  = blockIdx.x;           // 0..(64*256-1)
  const int b    = bvp >> 8;
  const int vp   = bvp & 255;            // v-pair index
  const int tid  = threadIdx.x;
  const int lane = tid & 31;
  const int wave = tid >> 5;             // 0..7
  const int par  = wave >> 2;            // v parity
  const int ot   = wave & 3;             // o tile
  const int half = lane >> 4;
  const int l15  = lane & 15;

  const size_t blkStride = (size_t)BATCH * NND * CT;   // elements per H block
  const size_t base      = (size_t)(b * 256 + vp) * CT2;

  // stage 5 contiguous 3072B interleaved panels into LDS (192 uint4 each)
  for (int i = tid; i < 5 * 192; i += 256) {
    const int blk = i / 192, q = i % 192;
    ((uint4*)Hs)[blk * 192 + q] = ((const uint4*)(H + blk * blkStride + base))[q];
  }
  __syncthreads();

  v8f acc = (v8f){0.f,0.f,0.f,0.f,0.f,0.f,0.f,0.f};

#pragma unroll
  for (int kb = 0; kb < 320; kb += 32) {
    FragU a, bb;
    const int o = ot * 16 + l15;
#pragma unroll
    for (int r = 0; r < 8; ++r) {       // A: W[o][f], f-pairs contiguous
      const int f = kb + half * 16 + r * 2;
      a.u[r] = *(const uint32_t*)(Wb + o * 320 + f);
    }
#pragma unroll
    for (int r = 0; r < 8; ++r) {       // B: H[f][t] (this parity), pack f,f+1
      const int f = kb + (half * 8 + r) * 2;
      uint32_t lo = Hs[(f * TT + l15) * 2 + par];
      uint32_t hi = Hs[((f + 1) * TT + l15) * 2 + par];
      bb.u[r] = lo | (hi << 16);
    }
    acc = __builtin_amdgcn_wmma_f32_16x16x32_bf16(
        false, a.b, false, bb.b, (short)0, acc, false, false);
  }

  if (l15 < TT) {                        // drop padded t columns
    const int v = vp * 2 + par;
#pragma unroll
    for (int r = 0; r < 8; ++r) {
      const int o = ot * 16 + half * 8 + r;
      out[(((size_t)b * 64 + o) * NND + v) * TT + l15] = acc[r] + bias[o];
    }
  }
}

// -------------------------------------------------------------------
extern "C" void kernel_launch(void* const* d_in, const int* in_sizes, int n_in,
                              void* d_out, int out_size, void* d_ws, size_t ws_size,
                              hipStream_t stream) {
  (void)in_sizes; (void)n_in; (void)out_size; (void)ws_size;

  const float* x    = (const float*)d_in[0];   // [64,64,512,12]
  const float* sup  = (const float*)d_in[1];   // [2,512,512]
  const float* W    = (const float*)d_in[2];   // [64,320]
  const float* bias = (const float*)d_in[3];   // [64]
  float* out = (float*)d_out;

  uint16_t* ws = (uint16_t*)d_ws;
  const size_t blk = (size_t)BATCH * NND * CT;  // 25,165,824 bf16 elements
  uint16_t* Xpk = ws;                           // H block 0 (x itself)
  uint16_t* Z10 = ws + 1 * blk;                 // H block 1 (A0 x)
  uint16_t* Z20 = ws + 2 * blk;                 // H block 2 (A0^2 x)
  uint16_t* Z11 = ws + 3 * blk;                 // H block 3 (A1 x)
  uint16_t* Z21 = ws + 4 * blk;                 // H block 4 (A1^2 x)
  uint16_t* AT  = ws + 5 * blk;                 // [2][512][512]
  uint16_t* Wb  = AT + 2 * NND * NND;           // [64][320]

  convert_x_kernel<<<(int)(blk / 256), 256, 0, stream>>>(x, Xpk);
  convert_aw_kernel<<<(2 * NND * NND + 64 * 320) / 256, 256, 0, stream>>>(sup, W, AT, Wb);

  dim3 g(CT / 128, NND / 128, BATCH);           // (6, 4, 64)
  diffusion_gemm_kernel<<<g, 256, 0, stream>>>(AT,             Xpk, Z10);
  diffusion_gemm_kernel<<<g, 256, 0, stream>>>(AT,             Z10, Z20);
  diffusion_gemm_kernel<<<g, 256, 0, stream>>>(AT + NND * NND, Xpk, Z11);
  diffusion_gemm_kernel<<<g, 256, 0, stream>>>(AT + NND * NND, Z11, Z21);

  proj_gemm_kernel<<<BATCH * 256, 256, 0, stream>>>(Xpk, Wb, bias, out);
}
